// GNNPolicy_85968065397283
// MI455X (gfx1250) — compile-verified, matrix operand order checked by reading, not run on hardware
//
#include <hip/hip_runtime.h>
#include <stdint.h>

#define IN_DIM 128
#define HID    256
#define EMB    128

typedef __attribute__((ext_vector_type(16))) __bf16       v16bf;
typedef __attribute__((ext_vector_type(8)))  float        v8f;
typedef __attribute__((ext_vector_type(4)))  unsigned int uint4v;
typedef __attribute__((ext_vector_type(2)))  unsigned int uint2v;

// round-to-nearest-even f32 -> bf16 (as raw u16)
__device__ __forceinline__ unsigned short f2bf(float f) {
    unsigned int u = __float_as_uint(f);
    u += 0x7FFFu + ((u >> 16) & 1u);
    return (unsigned short)(u >> 16);
}

// ---------------------------------------------------------------- utilities
__global__ void fill_kernel(float* p, float v, int n) {
    int i = blockIdx.x * blockDim.x + threadIdx.x;
    if (i < n) p[i] = v;
}

__global__ void deg_kernel(const int* __restrict__ ei, float* deg, int E) {
    int e = blockIdx.x * blockDim.x + threadIdx.x;
    if (e < E) atomicAdd(&deg[ei[E + e]], 1.0f);
}

__global__ void rsqrt_kernel(float* d, int n) {
    int i = blockIdx.x * blockDim.x + threadIdx.x;
    if (i < n) d[i] = rsqrtf(d[i]);
}

// ---------------------------------------------------------------- WMMA GEMM
// C[M,Nc] = A[M,K] @ W[K,Nc], all f32 in memory, bf16 operands, f32 accum.
// Block: 256 threads (8 waves). Block tile: 16 (M) x 128 (N). K chunk: 64.
// M % 16 == 0, K % 64 == 0, Nc % 128 == 0 (true for all call sites).
#define BK  64
#define BN  128
#define LDP 72   // padded LDS stride (elements); keeps 16B alignment, avoids bank conflicts

__global__ __launch_bounds__(256) void wmma_gemm_kernel(
    const float* __restrict__ A, const float* __restrict__ W,
    float* __restrict__ C, int K, int Nc) {
    __shared__ __align__(16) unsigned short sA[16 * LDP];  // A tile, row-major [m][k]
    __shared__ __align__(16) unsigned short sB[BN * LDP];  // B tile, transposed [n][k]

    const int tid   = threadIdx.x;
    const int mBase = blockIdx.x * 16;
    const int nBase = blockIdx.y * BN;
    const int wave  = tid >> 5;
    const int lane  = tid & 31;
    const int mn    = lane & 15;   // row of A / col of B owned by this lane
    const int khalf = lane >> 4;   // lanes 0-15: K {0..7,16..23}; 16-31: {8..15,24..31}

    // A-staging coordinates: one float4 per thread covers the whole 16x64 tile
    const int ar  = tid >> 4;          // row 0..15
    const int ac4 = (tid & 15) << 2;   // col 0,4,...,60

    v8f acc = {0.f, 0.f, 0.f, 0.f, 0.f, 0.f, 0.f, 0.f};

    for (int kc = 0; kc < K; kc += BK) {
        __syncthreads();

        // ---- stage A (16 x 64): 1 coalesced float4 load + 1 packed b64 store
        {
            const float4 av = *(const float4*)(&A[(size_t)(mBase + ar) * K + kc + ac4]);
            unsigned int lo = (unsigned int)f2bf(av.x) | ((unsigned int)f2bf(av.y) << 16);
            unsigned int hi = (unsigned int)f2bf(av.z) | ((unsigned int)f2bf(av.w) << 16);
            *(uint2v*)(&sA[ar * LDP + ac4]) = (uint2v){lo, hi};
        }

        // ---- stage B (64 x 128) transposed into [n][k]
        // issue all 8 coalesced float4 loads first so they overlap, then scatter
        float4 wv[8];
#pragma unroll
        for (int it = 0; it < 8; ++it) {
            int idx = tid + it * 256;
            int k = idx >> 5, n4 = (idx & 31) << 2;
            wv[it] = *(const float4*)(&W[(size_t)(kc + k) * Nc + nBase + n4]);
        }
#pragma unroll
        for (int it = 0; it < 8; ++it) {
            int idx = tid + it * 256;
            int k = idx >> 5, n4 = (idx & 31) << 2;
            unsigned short* bp = &sB[n4 * LDP + k];
            bp[0 * LDP] = f2bf(wv[it].x);
            bp[1 * LDP] = f2bf(wv[it].y);
            bp[2 * LDP] = f2bf(wv[it].z);
            bp[3 * LDP] = f2bf(wv[it].w);
        }
        __syncthreads();

#pragma unroll
        for (int k0 = 0; k0 < BK; k0 += 32) {
            union { v16bf v; uint4v q[2]; } fa, fb;
            const unsigned short* pa = sA + mn * LDP + k0 + khalf * 8;
            fa.q[0] = *(const uint4v*)(pa);        // K = k0 + khalf*8 + 0..7
            fa.q[1] = *(const uint4v*)(pa + 16);   // K = k0 + khalf*8 + 16..23
            const unsigned short* pb = sB + (wave * 16 + mn) * LDP + k0 + khalf * 8;
            fb.q[0] = *(const uint4v*)(pb);
            fb.q[1] = *(const uint4v*)(pb + 16);
            acc = __builtin_amdgcn_wmma_f32_16x16x32_bf16(
                false, fa.v, false, fb.v, (short)0, acc, false, false);
        }
    }

    // C/D layout: VGPR r -> row mBase + khalf*8 + r, col = lane&15 within wave tile
    const int row0 = mBase + khalf * 8;
    const int col  = nBase + wave * 16 + mn;
#pragma unroll
    for (int r = 0; r < 8; ++r)
        C[(size_t)(row0 + r) * Nc + col] = acc[r];
}

// ---------------------------------------------------------------- GCN aggregation
// AGG[i,:] = H[i,:] * dinv[i]^2   (self-loop term seeds the scatter target)
__global__ void selfterm_kernel(const float* __restrict__ H, const float* __restrict__ dinv,
                                float* __restrict__ AGG, int total, int ldShift) {
    int t = blockIdx.x * blockDim.x + threadIdx.x;
    if (t >= total) return;
    float di = dinv[t >> ldShift];
    AGG[t] = H[t] * di * di;
}

// AGG[dst,:] += H[src,:] * dinv[src]*dinv[dst]; one thread = one edge x 4 features
__global__ void scatter_kernel(const float* __restrict__ H, float* AGG,
                               const int* __restrict__ ei, const float* __restrict__ dinv,
                               int E, int vecShift, int Nc) {
    int t = blockIdx.x * blockDim.x + threadIdx.x;
    int e = t >> vecShift;
    if (e >= E) return;
    int j = (t & ((1 << vecShift) - 1)) << 2;
    int s = ei[e], d = ei[E + e];
    float nrm = dinv[s] * dinv[d];
    const float4 hv = *(const float4*)(H + (size_t)s * Nc + j);
    float* ap = AGG + (size_t)d * Nc + j;
    atomicAdd(ap + 0, hv.x * nrm);
    atomicAdd(ap + 1, hv.y * nrm);
    atomicAdd(ap + 2, hv.z * nrm);
    atomicAdd(ap + 3, hv.w * nrm);
}

__global__ void bias_relu_kernel(float* AGG, const float* __restrict__ b, int total, int mask) {
    int t = blockIdx.x * blockDim.x + threadIdx.x;
    if (t < total) AGG[t] = fmaxf(AGG[t] + b[t & mask], 0.0f);
}

// ---------------------------------------------------------------- readout heads
// g[j] = mean_i h2[i][j]; one block per column j (EMB blocks)
__global__ void colmean_kernel(const float* __restrict__ h2, float* g, int N) {
    int j = blockIdx.x, t = threadIdx.x;
    float s = 0.f;
    for (int i = t; i < N; i += 256) s += h2[(size_t)i * EMB + j];
    __shared__ float red[256];
    red[t] = s; __syncthreads();
    for (int k = 128; k > 0; k >>= 1) {
        if (t < k) red[t] += red[t + k];
        __syncthreads();
    }
    if (t == 0) g[j] = red[0] / (float)N;
}

// node_logits[i,c] = h2[i,:] . np_W[:,c] + np_b[c]
__global__ void nodehead_kernel(const float* __restrict__ h2, const float* __restrict__ npW,
                                const float* __restrict__ npb, float* out, int N) {
    int t = blockIdx.x * blockDim.x + threadIdx.x;
    if (t >= N * 3) return;
    int i = t / 3, c = t - i * 3;
    const float* hr = h2 + (size_t)i * EMB;
    float acc = npb[c];
#pragma unroll 4
    for (int k = 0; k < EMB; ++k) acc += hr[k] * npW[k * 3 + c];
    out[t] = acc;
}

// p[i][half*2+c] = h2[i,:] . ep_W[half*EMB : (half+1)*EMB, c]
__global__ void edgeproj_kernel(const float* __restrict__ h2, const float* __restrict__ epW,
                                float* p, int N) {
    int t = blockIdx.x * blockDim.x + threadIdx.x;
    if (t >= N * 4) return;
    int i = t >> 2, q = t & 3;
    int half = q >> 1, c = q & 1;
    const float* hr = h2 + (size_t)i * EMB;
    const float* w  = epW + (size_t)half * EMB * 2 + c;
    float acc = 0.f;
#pragma unroll 4
    for (int k = 0; k < EMB; ++k) acc += hr[k] * w[k * 2];
    p[t] = acc;
}

// edge_logits[e,c] = p1[src][c] + p2[dst][c] + ep_b[c]
__global__ void edgehead_kernel(const float* __restrict__ p, const int* __restrict__ ei,
                                const float* __restrict__ epb, float* out, int E) {
    int e = blockIdx.x * blockDim.x + threadIdx.x;
    if (e >= E) return;
    int s = ei[e], d = ei[E + e];
    const float4 ps = *(const float4*)(p + (size_t)s * 4);
    const float4 pd = *(const float4*)(p + (size_t)d * 4);
    out[2 * e + 0] = ps.x + pd.z + epb[0];
    out[2 * e + 1] = ps.y + pd.w + epb[1];
}

// value = relu(g@fc1_W + fc1_b) @ fc2_W + fc2_b ; global = g@gp_W + gp_b
__global__ void mlphead_kernel(const float* __restrict__ g,
                               const float* __restrict__ fc1W, const float* __restrict__ fc1b,
                               const float* __restrict__ fc2W, const float* __restrict__ fc2b,
                               const float* __restrict__ gpW,  const float* __restrict__ gpb,
                               float* out_glob, float* out_val) {
    __shared__ float red[256];
    int t = threadIdx.x;
    float acc = fc1b[t];
#pragma unroll 4
    for (int k = 0; k < EMB; ++k) acc += g[k] * fc1W[k * HID + t];
    float v = fmaxf(acc, 0.f);
    red[t] = v * fc2W[t]; __syncthreads();
    for (int k = 128; k > 0; k >>= 1) {
        if (t < k) red[t] += red[t + k];
        __syncthreads();
    }
    if (t == 0) *out_val = red[0] + fc2b[0];
    __syncthreads();
    red[t] = (t < EMB) ? g[t] * gpW[t] : 0.f; __syncthreads();
    for (int k = 128; k > 0; k >>= 1) {
        if (t < k) red[t] += red[t + k];
        __syncthreads();
    }
    if (t == 0) *out_glob = red[0] + gpb[0];
}

// ---------------------------------------------------------------- launcher
extern "C" void kernel_launch(void* const* d_in, const int* in_sizes, int n_in,
                              void* d_out, int out_size, void* d_ws, size_t ws_size,
                              hipStream_t stream) {
    const float* x    = (const float*)d_in[0];
    const int*   ei   = (const int*)  d_in[1];
    const float* W1   = (const float*)d_in[2];
    const float* b1   = (const float*)d_in[3];
    const float* W2   = (const float*)d_in[4];
    const float* b2   = (const float*)d_in[5];
    const float* fc1W = (const float*)d_in[6];
    const float* fc1b = (const float*)d_in[7];
    const float* fc2W = (const float*)d_in[8];
    const float* fc2b = (const float*)d_in[9];
    const float* npW  = (const float*)d_in[10];
    const float* npb  = (const float*)d_in[11];
    const float* epW  = (const float*)d_in[12];
    const float* epb  = (const float*)d_in[13];
    const float* gpW  = (const float*)d_in[14];
    const float* gpb  = (const float*)d_in[15];

    const int N = in_sizes[0] / IN_DIM;   // 50000
    const int E = in_sizes[1] / 2;        // 800000

    float* ws = (float*)d_ws;
    size_t off = 0;
    auto alloc = [&](size_t n) { size_t o = off; off += (n + 63) & ~size_t(63); return o; };
    float* dinv = ws + alloc((size_t)N);
    float* g    = ws + alloc(256);
    float* pbuf = ws + alloc((size_t)N * 4);
    float* H1   = ws + alloc((size_t)N * HID);   // reused for H2 + h2 after layer 1
    float* h1   = ws + alloc((size_t)N * HID);
    float* H2   = H1;
    float* h2   = H1 + (size_t)N * EMB;

    const dim3 blk(256);
    auto nb = [](long long n) { return (unsigned)((n + 255) / 256); };

    // degree -> dinv (self-loop adds 1)
    fill_kernel<<<nb(N), blk, 0, stream>>>(dinv, 1.0f, N);
    deg_kernel<<<nb(E), blk, 0, stream>>>(ei, dinv, E);
    rsqrt_kernel<<<nb(N), blk, 0, stream>>>(dinv, N);

    // layer 1: H1 = x @ W1 ; h1 = relu(agg + b1)
    wmma_gemm_kernel<<<dim3(N / 16, HID / BN), blk, 0, stream>>>(x, W1, H1, IN_DIM, HID);
    selfterm_kernel<<<nb((long long)N * HID), blk, 0, stream>>>(H1, dinv, h1, N * HID, 8);
    scatter_kernel<<<nb((long long)E * (HID / 4)), blk, 0, stream>>>(H1, h1, ei, dinv, E, 6, HID);
    bias_relu_kernel<<<nb((long long)N * HID), blk, 0, stream>>>(h1, b1, N * HID, HID - 1);

    // layer 2: H2 = h1 @ W2 ; h2 = relu(agg + b2)
    wmma_gemm_kernel<<<dim3(N / 16, EMB / BN), blk, 0, stream>>>(h1, W2, H2, HID, EMB);
    selfterm_kernel<<<nb((long long)N * EMB), blk, 0, stream>>>(H2, dinv, h2, N * EMB, 7);
    scatter_kernel<<<nb((long long)E * (EMB / 4)), blk, 0, stream>>>(H2, h2, ei, dinv, E, 5, EMB);
    bias_relu_kernel<<<nb((long long)N * EMB), blk, 0, stream>>>(h2, b2, N * EMB, EMB - 1);

    // heads
    colmean_kernel<<<EMB, blk, 0, stream>>>(h2, g, N);

    float* out = (float*)d_out;
    nodehead_kernel<<<nb((long long)N * 3), blk, 0, stream>>>(h2, npW, npb, out, N);
    edgeproj_kernel<<<nb((long long)N * 4), blk, 0, stream>>>(h2, epW, pbuf, N);
    edgehead_kernel<<<nb(E), blk, 0, stream>>>(pbuf, ei, epb, out + (size_t)N * 3, E);
    mlphead_kernel<<<1, blk, 0, stream>>>(g, fc1W, fc1b, fc2W, fc2b, gpW, gpb,
                                          out + (size_t)N * 3 + (size_t)E * 2,
                                          out + (size_t)N * 3 + (size_t)E * 2 + 1);
}
